// PCN_encoder_25744033972950
// MI455X (gfx1250) — matrix operand order repaired and checked by reading, not compiled, not run on hardware
//
#include <hip/hip_runtime.h>
#include <hip/hip_bf16.h>

// ---------------------------------------------------------------------------
// PointNet++ style pipeline for MI455X (gfx1250, wave32, WMMA).
// All MLP einsums are bf16 WMMA GEMMs (V_WMMA_F32_16X16X32_BF16) with fused
// bias + ReLU. Irregular stages (FPS / ball query / kNN3 / grouping) are
// bandwidth-bound helper kernels that emit bf16 A-matrices padded to K%32==0.
// ---------------------------------------------------------------------------

typedef __attribute__((ext_vector_type(16))) __bf16 v16bf;
typedef __attribute__((ext_vector_type(8)))  float  v8f;
typedef __attribute__((ext_vector_type(8)))  unsigned short u16x8;

struct U16x16 { unsigned short d[16]; };

__device__ __forceinline__ unsigned short f2bf(float f) {
    unsigned int u = __float_as_uint(f);
    unsigned int r = u + 0x7FFFu + ((u >> 16) & 1u);   // round-to-nearest-even
    return (unsigned short)(r >> 16);
}
__device__ __forceinline__ float bf2f(unsigned short h) {
    return __uint_as_float(((unsigned int)h) << 16);
}

// ---------------------------------------------------------------------------
// Problem constants
// ---------------------------------------------------------------------------
#define BB   8
#define NN   8192
#define CC   24
#define S1   1024
#define KS1  16
#define S2   512
#define KS2  32
#define FPSB 1024

// ---------------------------------------------------------------------------
// 0) transpose (B,C,N) -> (B,N,C)
// ---------------------------------------------------------------------------
__global__ void transpose_kernel(const float* __restrict__ xyz, float* __restrict__ xt,
                                 long long total) {
    long long t = (long long)blockIdx.x * blockDim.x + threadIdx.x;
    if (t >= total) return;
    int c = (int)(t % CC);
    long long bn = t / CC;
    int n = (int)(bn % NN);
    int b = (int)(bn / NN);
    xt[t] = xyz[((long long)b * CC + c) * NN + n];
}

// ---------------------------------------------------------------------------
// 1) farthest point sampling. One workgroup per batch (serial argmax chain).
// ---------------------------------------------------------------------------
__global__ void __launch_bounds__(FPSB) fps_kernel(
    const float* __restrict__ coords, int ldc, int Npts, int npoint,
    int* __restrict__ out_idx, float* __restrict__ dist) {
    const int b   = blockIdx.x;
    const int tid = threadIdx.x;
    const float* base = coords + (long long)b * Npts * ldc;
    float* db = dist + (long long)b * Npts;

    __shared__ float sm[FPSB];
    __shared__ int   si[FPSB];
    __shared__ int   s_far;

    for (int n = tid; n < Npts; n += FPSB) db[n] = 1e10f;
    if (tid == 0) s_far = 0;
    __syncthreads();

    for (int it = 0; it < npoint; ++it) {
        const int far = s_far;
        if (tid == 0) out_idx[b * npoint + it] = far;
        const float cx = base[(long long)far * ldc + 0];
        const float cy = base[(long long)far * ldc + 1];
        const float cz = base[(long long)far * ldc + 2];
        float bm = -1.0f; int bi = Npts;
        for (int n = tid; n < Npts; n += FPSB) {
            float dx = base[(long long)n * ldc + 0] - cx;
            float dy = base[(long long)n * ldc + 1] - cy;
            float dz = base[(long long)n * ldc + 2] - cz;
            float d  = dx * dx + dy * dy + dz * dz;
            float od = db[n];
            d = d < od ? d : od;
            db[n] = d;
            if (d > bm || (d == bm && n < bi)) { bm = d; bi = n; }
        }
        sm[tid] = bm; si[tid] = bi;
        __syncthreads();
        for (int s = FPSB / 2; s > 0; s >>= 1) {
            if (tid < s) {
                float a = sm[tid], c2 = sm[tid + s];
                if (c2 > a || (c2 == a && si[tid + s] < si[tid])) {
                    sm[tid] = c2; si[tid] = si[tid + s];
                }
            }
            __syncthreads();
        }
        if (tid == 0) s_far = si[0];
        __syncthreads();
    }
}

// ---------------------------------------------------------------------------
// 2) gather sampled xyz (packed stride-3 output)
// ---------------------------------------------------------------------------
__global__ void gather_xyz_kernel(const float* __restrict__ coords, int ldc, int Npts,
                                  const int* __restrict__ fidx, int S,
                                  float* __restrict__ out, long long total) {
    long long t = (long long)blockIdx.x * blockDim.x + threadIdx.x;
    if (t >= total) return;
    int c = (int)(t % 3);
    long long bs = t / 3;
    int s = (int)(bs % S);
    int b = (int)(bs / S);
    int n = fidx[b * S + s];
    out[t] = coords[((long long)b * Npts + n) * ldc + c];
}

// ---------------------------------------------------------------------------
// 3) ball query: first nsample indices (ascending n) within radius
// ---------------------------------------------------------------------------
__global__ void ball_query_kernel(const float* __restrict__ coords, int ldc, int Npts,
                                  const float* __restrict__ q, int S, float r2,
                                  int nsample, int* __restrict__ out, long long total) {
    long long t = (long long)blockIdx.x * blockDim.x + threadIdx.x;
    if (t >= total) return;
    int s = (int)(t % S);
    int b = (int)(t / S);
    const float* base = coords + (long long)b * Npts * ldc;
    const float qx = q[((long long)b * S + s) * 3 + 0];
    const float qy = q[((long long)b * S + s) * 3 + 1];
    const float qz = q[((long long)b * S + s) * 3 + 2];
    int* o = out + ((long long)b * S + s) * nsample;
    int cnt = 0, first = -1;
    for (int n = 0; n < Npts; ++n) {
        float dx = base[(long long)n * ldc + 0] - qx;
        float dy = base[(long long)n * ldc + 1] - qy;
        float dz = base[(long long)n * ldc + 2] - qz;
        float d2 = dx * dx + dy * dy + dz * dz;
        if (d2 <= r2) {
            if (first < 0) first = n;
            o[cnt++] = n;
            if (cnt == nsample) break;
        }
    }
    if (first < 0) first = 0;
    for (int k = cnt; k < nsample; ++k) o[k] = first;
}

// ---------------------------------------------------------------------------
// 4) grouping: build bf16 A matrix rows=(b,s,k), cols=[rel_xyz(3) | feats | pad]
// ---------------------------------------------------------------------------
__global__ void group_kernel(const float* __restrict__ coords, int ldc,
                             const float* __restrict__ feats, int ldf, int fdim,
                             const float* __restrict__ new_xyz,
                             const int* __restrict__ idx,
                             int Npts, int S, int K, int Kpad,
                             unsigned short* __restrict__ A, long long total) {
    long long t = (long long)blockIdx.x * blockDim.x + threadIdx.x;
    if (t >= total) return;
    int c = (int)(t % Kpad);
    long long row = t / Kpad;          // (b*S+s)*K + k
    long long bs  = row / K;
    int b = (int)(bs / S);
    int n = idx[row];
    float v = 0.0f;
    if (c < 3) {
        v = coords[((long long)b * Npts + n) * ldc + c] - new_xyz[bs * 3 + c];
    } else if (c < 3 + fdim) {
        v = feats[((long long)b * Npts + n) * ldf + (c - 3)];
    }
    A[t] = f2bf(v);
}

// ---------------------------------------------------------------------------
// 5) weight conversion: W (O,C) f32 -> Bmat (O,Kpad) bf16, row-major.
//    W's natural (o,c) layout IS the [n][k] layout the GEMM's LDS tile wants,
//    so no transpose is ever needed in the GEMM inner loop.
// ---------------------------------------------------------------------------
__global__ void wcvt_kernel(const float* __restrict__ W, int O, int C, int Kpad,
                            unsigned short* __restrict__ Bm, long long total) {
    long long t = (long long)blockIdx.x * blockDim.x + threadIdx.x;
    if (t >= total) return;
    int k = (int)(t % Kpad);
    int o = (int)(t / Kpad);
    Bm[t] = (k < C) ? f2bf(W[(long long)o * C + k]) : (unsigned short)0;
}

// ---------------------------------------------------------------------------
// 6) WMMA GEMM: out = relu(A(rows,Kpad) * W^T + bias), W given as Bm(O,Kpad).
//    64x64 block tile, 8 waves, each wave computes a 16x32 strip:
//    two v_wmma_f32_16x16x32_bf16 per K-step sharing one A fragment.
//    Both LDS tiles are fragment-major, so staging is 1 global_load_b128 +
//    1 ds_store_b128 per thread per tile, and every fragment is 2x
//    ds_load_b128 (A and B both use contiguous k = 16*half + s per lane):
//      As[m][k] row-major, Bt[n][k] row-major (= W's own layout)
// ---------------------------------------------------------------------------
__global__ void __launch_bounds__(256) gemm_bias_relu_wmma(
    const unsigned short* __restrict__ A, int lda,
    const unsigned short* __restrict__ Bm, int ldb,   // ldb == Kpad
    const float* __restrict__ bias, int Kpad, int O,
    unsigned short* __restrict__ outb, float* __restrict__ outf) {
    __shared__ __align__(16) unsigned short As[64][40];   // [m][k], 80B row pitch
    __shared__ __align__(16) unsigned short Bt[64][40];   // [n][k]

    const int tid  = threadIdx.x;
    const int lane = tid & 31;
    const int wave = tid >> 5;
    const int wm   = wave & 3;          // M subtile 0..3
    const int wn   = wave >> 2;         // N strip  0..1 (each 32 wide)
    const long long blockM = (long long)blockIdx.x * 64;
    const int blockN = blockIdx.y * 64;

    const int mrow = lane & 15;
    const int half = lane >> 4;

    v8f acc0 = {}, acc1 = {};

    // staging coordinates (16B aligned: lda/ldb % 8 == 0, kb % 32 == 0)
    const int srow = tid >> 2;          // 0..63
    const int scol = (tid & 3) * 8;     // 0,8,16,24

    for (int kb = 0; kb < Kpad; kb += 32) {
        // prefetch next A K-tile (global_prefetch_b8)
        if (kb + 32 < Kpad)
            __builtin_prefetch(&A[(blockM + srow) * (long long)lda + kb + 32], 0, 0);

        // A tile 64x32: one global_load_b128 + one ds_store_b128 per thread
        u16x8 av8 = *(const u16x8*)&A[(blockM + srow) * (long long)lda + kb + scol];
        *(u16x8*)&As[srow][scol] = av8;

        // B tile 64x32 (rows = output channels): same pattern, no transpose
        u16x8 bv8 = *(const u16x8*)&Bm[(long long)(blockN + srow) * ldb + kb + scol];
        *(u16x8*)&Bt[srow][scol] = bv8;
        __syncthreads();

        // fragments: 2x ds_load_b128 each (k = 16*half + s per lane, same for A and B)
        const int am = wm * 16 + mrow;
        U16x16 au, bu0, bu1;
        *(u16x8*)&au.d[0] = *(const u16x8*)&As[am][16 * half];
        *(u16x8*)&au.d[8] = *(const u16x8*)&As[am][16 * half + 8];
        const int bn0 = wn * 32 + mrow;
        *(u16x8*)&bu0.d[0] = *(const u16x8*)&Bt[bn0][16 * half];
        *(u16x8*)&bu0.d[8] = *(const u16x8*)&Bt[bn0][16 * half + 8];
        *(u16x8*)&bu1.d[0] = *(const u16x8*)&Bt[bn0 + 16][16 * half];
        *(u16x8*)&bu1.d[8] = *(const u16x8*)&Bt[bn0 + 16][16 * half + 8];

        v16bf va  = __builtin_bit_cast(v16bf, au);
        v16bf vb0 = __builtin_bit_cast(v16bf, bu0);
        v16bf vb1 = __builtin_bit_cast(v16bf, bu1);
        acc0 = __builtin_amdgcn_wmma_f32_16x16x32_bf16(
            false, va, false, vb0, (short)0, acc0, false, false);
        acc1 = __builtin_amdgcn_wmma_f32_16x16x32_bf16(
            false, va, false, vb1, (short)0, acc1, false, false);
        __syncthreads();
    }

    // epilogue: bias + ReLU; D layout: VGPR r -> M = r + 8*half, N = lane%16
    const int nCol0 = blockN + wn * 32 + mrow;
    const int nCol1 = nCol0 + 16;
    const float bb0 = bias[nCol0];
    const float bb1 = bias[nCol1];
#pragma unroll
    for (int r = 0; r < 8; ++r) {
        long long row = blockM + wm * 16 + (r + 8 * half);
        float v0 = acc0[r] + bb0; v0 = v0 > 0.0f ? v0 : 0.0f;
        float v1 = acc1[r] + bb1; v1 = v1 > 0.0f ? v1 : 0.0f;
        if (outb) {
            outb[row * O + nCol0] = f2bf(v0);
            outb[row * O + nCol1] = f2bf(v1);
        }
        if (outf) {
            outf[row * O + nCol0] = v0;
            outf[row * O + nCol1] = v1;
        }
    }
}

// ---------------------------------------------------------------------------
// 7) max-pool over the nsample axis (bf16 act -> f32 features)
// ---------------------------------------------------------------------------
__global__ void maxpool_kernel(const unsigned short* __restrict__ act, int O, int K,
                               float* __restrict__ out, long long total) {
    long long t = (long long)blockIdx.x * blockDim.x + threadIdx.x;
    if (t >= total) return;
    int o = (int)(t % O);
    long long bs = t / O;
    float m = -1e30f;
    for (int k = 0; k < K; ++k) {
        float v = bf2f(act[(bs * K + k) * (long long)O + o]);
        m = v > m ? v : m;
    }
    out[t] = m;
}

// ---------------------------------------------------------------------------
// 8) 3-NN with inverse-distance weights
// ---------------------------------------------------------------------------
__global__ void knn3_kernel(const float* __restrict__ xyz1, int ld1, int N1,
                            const float* __restrict__ xyz2, int ld2, int N2,
                            int* __restrict__ oidx, float* __restrict__ ow,
                            long long total) {
    long long t = (long long)blockIdx.x * blockDim.x + threadIdx.x;
    if (t >= total) return;
    int n = (int)(t % N1);
    int b = (int)(t / N1);
    const float px = xyz1[((long long)b * N1 + n) * ld1 + 0];
    const float py = xyz1[((long long)b * N1 + n) * ld1 + 1];
    const float pz = xyz1[((long long)b * N1 + n) * ld1 + 2];
    const float* base = xyz2 + (long long)b * N2 * ld2;
    float d0 = 1e30f, d1 = 1e30f, d2 = 1e30f;
    int   i0 = 0,     i1 = 0,     i2 = 0;
    for (int m = 0; m < N2; ++m) {
        float dx = base[(long long)m * ld2 + 0] - px;
        float dy = base[(long long)m * ld2 + 1] - py;
        float dz = base[(long long)m * ld2 + 2] - pz;
        float d = dx * dx + dy * dy + dz * dz;
        if (d < d0)      { d2 = d1; i2 = i1; d1 = d0; i1 = i0; d0 = d; i0 = m; }
        else if (d < d1) { d2 = d1; i2 = i1; d1 = d;  i1 = m; }
        else if (d < d2) { d2 = d;  i2 = m; }
    }
    float r0 = 1.0f / (d0 + 1e-8f);
    float r1 = 1.0f / (d1 + 1e-8f);
    float r2 = 1.0f / (d2 + 1e-8f);
    float inv = 1.0f / (r0 + r1 + r2);
    oidx[t * 3 + 0] = i0; oidx[t * 3 + 1] = i1; oidx[t * 3 + 2] = i2;
    ow[t * 3 + 0] = r0 * inv; ow[t * 3 + 1] = r1 * inv; ow[t * 3 + 2] = r2 * inv;
}

// ---------------------------------------------------------------------------
// 9) interpolate + concat -> bf16 A matrix for FP MLPs
//    mode 0: feat1 = dense f32 (ldf1, d1 dims)
//    mode 1: feat1 = concat(l0_xyz, x) pulled from xt (27 dims, ld 24)
// ---------------------------------------------------------------------------
__global__ void interp_concat_kernel(const float* __restrict__ feat1, int ldf1, int d1,
                                     int mode, const float* __restrict__ xt,
                                     const float* __restrict__ pts2, int ldp2, int d2,
                                     int np2, const int* __restrict__ kidx,
                                     const float* __restrict__ kw,
                                     int Npts1, int Kpad,
                                     unsigned short* __restrict__ A, long long total) {
    long long t = (long long)blockIdx.x * blockDim.x + threadIdx.x;
    if (t >= total) return;
    int c = (int)(t % Kpad);
    long long row = t / Kpad;          // b*Npts1 + n
    int b = (int)(row / Npts1);
    float v = 0.0f;
    if (c < d1) {
        if (mode == 0) {
            v = feat1[row * (long long)ldf1 + c];
        } else {
            int cc = (c < 3) ? c : (c - 3);
            v = xt[row * (long long)CC + cc];
        }
    } else if (c < d1 + d2) {
        int j = c - d1;
        const int*   ki = kidx + row * 3;
        const float* wv = kw   + row * 3;
        v = wv[0] * pts2[((long long)b * np2 + ki[0]) * ldp2 + j]
          + wv[1] * pts2[((long long)b * np2 + ki[1]) * ldp2 + j]
          + wv[2] * pts2[((long long)b * np2 + ki[2]) * ldp2 + j];
    }
    A[t] = f2bf(v);
}

// ---------------------------------------------------------------------------
// Host side
// ---------------------------------------------------------------------------
static inline long long cdiv(long long a, long long b) { return (a + b - 1) / b; }

extern "C" void kernel_launch(void* const* d_in, const int* in_sizes, int n_in,
                              void* d_out, int out_size, void* d_ws, size_t ws_size,
                              hipStream_t stream) {
    const float* xyz    = (const float*)d_in[0];
    const float* sa1_w0 = (const float*)d_in[1];  const float* sa1_b0 = (const float*)d_in[2];
    const float* sa1_w1 = (const float*)d_in[3];  const float* sa1_b1 = (const float*)d_in[4];
    const float* sa1_w2 = (const float*)d_in[5];  const float* sa1_b2 = (const float*)d_in[6];
    const float* sa2_w0 = (const float*)d_in[7];  const float* sa2_b0 = (const float*)d_in[8];
    const float* sa2_w1 = (const float*)d_in[9];  const float* sa2_b1 = (const float*)d_in[10];
    const float* fp2_w0 = (const float*)d_in[11]; const float* fp2_b0 = (const float*)d_in[12];
    const float* fp2_w1 = (const float*)d_in[13]; const float* fp2_b1 = (const float*)d_in[14];
    const float* fp1_w0 = (const float*)d_in[15]; const float* fp1_b0 = (const float*)d_in[16];
    const float* fp1_w1 = (const float*)d_in[17]; const float* fp1_b1 = (const float*)d_in[18];

    char* w = (char*)d_ws;
    size_t off = 0;
    auto take = [&](size_t bytes) { size_t o = off; off = (off + bytes + 255) & ~(size_t)255; return o; };

    float* xt      = (float*)(w + take((size_t)BB * NN * CC * 4));      // (B,N,24)
    float* dist    = (float*)(w + take((size_t)BB * NN * 4));
    int*   fps1    = (int*)  (w + take((size_t)BB * S1 * 4));
    float* l1xyz   = (float*)(w + take((size_t)BB * S1 * 3 * 4));
    int*   idx1    = (int*)  (w + take((size_t)BB * S1 * KS1 * 4));
    int*   fps2    = (int*)  (w + take((size_t)BB * S2 * 4));
    float* l2xyz   = (float*)(w + take((size_t)BB * S2 * 3 * 4));
    int*   idx2    = (int*)  (w + take((size_t)BB * S2 * KS2 * 4));
    float* l1pts   = (float*)(w + take((size_t)BB * S1 * 128 * 4));
    float* l2pts   = (float*)(w + take((size_t)BB * S2 * 256 * 4));
    float* l1new   = (float*)(w + take((size_t)BB * S1 * 256 * 4));
    int*   knni    = (int*)  (w + take((size_t)BB * NN * 3 * 4));
    float* knnw    = (float*)(w + take((size_t)BB * NN * 3 * 4));
    unsigned short* Bmat = (unsigned short*)(w + take((size_t)384 * 256 * 2));
    unsigned short* Abuf = (unsigned short*)(w + take((size_t)131072 * 160 * 2)); // 42 MB max
    unsigned short* act0 = (unsigned short*)(w + take((size_t)131072 * 128 * 2));
    unsigned short* act1 = (unsigned short*)(w + take((size_t)131072 * 256 * 2));
    (void)ws_size; (void)in_sizes; (void)n_in; (void)out_size;

    const int TPB = 256;
    long long tot;

    // ---- x = transpose(xyz) -------------------------------------------------
    tot = (long long)BB * NN * CC;
    transpose_kernel<<<cdiv(tot, TPB), TPB, 0, stream>>>(xyz, xt, tot);

    // ======================= SA1 ============================================
    fps_kernel<<<BB, FPSB, 0, stream>>>(xt, CC, NN, S1, fps1, dist);
    tot = (long long)BB * S1 * 3;
    gather_xyz_kernel<<<cdiv(tot, TPB), TPB, 0, stream>>>(xt, CC, NN, fps1, S1, l1xyz, tot);
    tot = (long long)BB * S1;
    ball_query_kernel<<<cdiv(tot, TPB), TPB, 0, stream>>>(
        xt, CC, NN, l1xyz, S1, 0.035f * 0.035f, KS1, idx1, tot);
    tot = (long long)BB * S1 * KS1 * 32;     // Kpad = 32 (27 -> 32)
    group_kernel<<<cdiv(tot, TPB), TPB, 0, stream>>>(
        xt, CC, xt, CC, CC, l1xyz, idx1, NN, S1, KS1, 32, Abuf, tot);

    const long long rows1 = (long long)BB * S1 * KS1;   // 131072
    // layer 0: 32 -> 64
    tot = (long long)64 * 32;
    wcvt_kernel<<<cdiv(tot, TPB), TPB, 0, stream>>>(sa1_w0, 64, 27, 32, Bmat, tot);
    gemm_bias_relu_wmma<<<dim3(rows1 / 64, 64 / 64), 256, 0, stream>>>(
        Abuf, 32, Bmat, 32, sa1_b0, 32, 64, act0, nullptr);
    // layer 1: 64 -> 128
    tot = (long long)128 * 64;
    wcvt_kernel<<<cdiv(tot, TPB), TPB, 0, stream>>>(sa1_w1, 128, 64, 64, Bmat, tot);
    gemm_bias_relu_wmma<<<dim3(rows1 / 64, 128 / 64), 256, 0, stream>>>(
        act0, 64, Bmat, 64, sa1_b1, 64, 128, act1, nullptr);
    // layer 2: 128 -> 128
    tot = (long long)128 * 128;
    wcvt_kernel<<<cdiv(tot, TPB), TPB, 0, stream>>>(sa1_w2, 128, 128, 128, Bmat, tot);
    gemm_bias_relu_wmma<<<dim3(rows1 / 64, 128 / 64), 256, 0, stream>>>(
        act1, 128, Bmat, 128, sa1_b2, 128, 128, act0, nullptr);
    // max-pool over K=16 -> l1_points (B,S1,128)
    tot = (long long)BB * S1 * 128;
    maxpool_kernel<<<cdiv(tot, TPB), TPB, 0, stream>>>(act0, 128, KS1, l1pts, tot);

    // ======================= SA2 ============================================
    fps_kernel<<<BB, FPSB, 0, stream>>>(l1xyz, 3, S1, S2, fps2, dist);
    tot = (long long)BB * S2 * 3;
    gather_xyz_kernel<<<cdiv(tot, TPB), TPB, 0, stream>>>(l1xyz, 3, S1, fps2, S2, l2xyz, tot);
    tot = (long long)BB * S2;
    ball_query_kernel<<<cdiv(tot, TPB), TPB, 0, stream>>>(
        l1xyz, 3, S1, l2xyz, S2, 0.0176f * 0.0176f, KS2, idx2, tot);
    tot = (long long)BB * S2 * KS2 * 160;    // Kpad = 160 (3+128=131 -> 160)
    group_kernel<<<cdiv(tot, TPB), TPB, 0, stream>>>(
        l1xyz, 3, l1pts, 128, 128, l2xyz, idx2, S1, S2, KS2, 160, Abuf, tot);

    const long long rows2 = (long long)BB * S2 * KS2;   // 131072
    // layer 0: 160 -> 128
    tot = (long long)128 * 160;
    wcvt_kernel<<<cdiv(tot, TPB), TPB, 0, stream>>>(sa2_w0, 128, 131, 160, Bmat, tot);
    gemm_bias_relu_wmma<<<dim3(rows2 / 64, 128 / 64), 256, 0, stream>>>(
        Abuf, 160, Bmat, 160, sa2_b0, 160, 128, act0, nullptr);
    // layer 1: 128 -> 256
    tot = (long long)256 * 128;
    wcvt_kernel<<<cdiv(tot, TPB), TPB, 0, stream>>>(sa2_w1, 256, 128, 128, Bmat, tot);
    gemm_bias_relu_wmma<<<dim3(rows2 / 64, 256 / 64), 256, 0, stream>>>(
        act0, 128, Bmat, 128, sa2_b1, 128, 256, act1, nullptr);
    // max-pool over K=32 -> l2_points (B,S2,256)
    tot = (long long)BB * S2 * 256;
    maxpool_kernel<<<cdiv(tot, TPB), TPB, 0, stream>>>(act1, 256, KS2, l2pts, tot);

    // ======================= FP2 (l2 -> l1) =================================
    tot = (long long)BB * S1;
    knn3_kernel<<<cdiv(tot, TPB), TPB, 0, stream>>>(l1xyz, 3, S1, l2xyz, 3, S2,
                                                    knni, knnw, tot);
    tot = (long long)BB * S1 * 384;          // Kpad = 384 (128+256)
    interp_concat_kernel<<<cdiv(tot, TPB), TPB, 0, stream>>>(
        l1pts, 128, 128, 0, nullptr, l2pts, 256, 256, S2, knni, knnw,
        S1, 384, Abuf, tot);

    const long long rowsF2 = (long long)BB * S1;        // 8192
    tot = (long long)256 * 384;
    wcvt_kernel<<<cdiv(tot, TPB), TPB, 0, stream>>>(fp2_w0, 256, 384, 384, Bmat, tot);
    gemm_bias_relu_wmma<<<dim3(rowsF2 / 64, 256 / 64), 256, 0, stream>>>(
        Abuf, 384, Bmat, 384, fp2_b0, 384, 256, act0, nullptr);
    tot = (long long)256 * 256;
    wcvt_kernel<<<cdiv(tot, TPB), TPB, 0, stream>>>(fp2_w1, 256, 256, 256, Bmat, tot);
    gemm_bias_relu_wmma<<<dim3(rowsF2 / 64, 256 / 64), 256, 0, stream>>>(
        act0, 256, Bmat, 256, fp2_b1, 256, 256, nullptr, l1new);

    // ======================= FP1 (l1 -> l0) =================================
    tot = (long long)BB * NN;
    knn3_kernel<<<cdiv(tot, TPB), TPB, 0, stream>>>(xt, CC, NN, l1xyz, 3, S1,
                                                    knni, knnw, tot);
    tot = (long long)BB * NN * 288;          // Kpad = 288 (27+256=283 -> 288)
    interp_concat_kernel<<<cdiv(tot, TPB), TPB, 0, stream>>>(
        nullptr, 0, 27, 1, xt, l1new, 256, 256, S1, knni, knnw,
        NN, 288, Abuf, tot);

    const long long rowsF1 = (long long)BB * NN;        // 65536
    tot = (long long)256 * 288;
    wcvt_kernel<<<cdiv(tot, TPB), TPB, 0, stream>>>(fp1_w0, 256, 283, 288, Bmat, tot);
    gemm_bias_relu_wmma<<<dim3(rowsF1 / 64, 256 / 64), 256, 0, stream>>>(
        Abuf, 288, Bmat, 288, fp1_b0, 288, 256, act0, nullptr);
    tot = (long long)256 * 256;
    wcvt_kernel<<<cdiv(tot, TPB), TPB, 0, stream>>>(fp1_w1, 256, 256, 256, Bmat, tot);
    gemm_bias_relu_wmma<<<dim3(rowsF1 / 64, 256 / 64), 256, 0, stream>>>(
        act0, 256, Bmat, 256, fp1_b1, 256, 256, nullptr, (float*)d_out);
}
